// NPairLoss_65429531787296
// MI455X (gfx1250) — matrix operand order repaired and checked by reading, not compiled, not run on hardware
//
#include <hip/hip_runtime.h>
#include <hip/hip_bf16.h>

typedef _Float16 half8  __attribute__((ext_vector_type(8)));
typedef _Float16 half16 __attribute__((ext_vector_type(16)));
typedef float    float8 __attribute__((ext_vector_type(8)));

#define DIM   256
#define NCOL  16384
#define NTILE (NCOL / 16)          // 1024 tiles per side
#define JCHUNKS 8                  // j-stride per wave; grid.y
#define EPS   1e-8f
#define LOG2E 1.44269504088896340736f
#define LN2   0.69314718055994530942f

#define ROWB      512              // bytes per packed row in global T
#define LROWB     528              // padded LDS row stride (16B pad -> banks 4r..)
#define LROWH     (LROWB / 2)      // halves per LDS row

// ---------------- Stage 1: column norms -> packed f16 transposed matrix ----------------
__global__ void norm_pack_kernel(const float* __restrict__ w,
                                 _Float16* __restrict__ T,
                                 float* __restrict__ rowS) {
    int n = blockIdx.x * blockDim.x + threadIdx.x;   // coalesced over n
    float ss = 0.0f;
    for (int d = 0; d < DIM; ++d) {
        float v = w[d * NCOL + n];
        ss += v * v;
    }
    float inv = 1.0f / fmaxf(sqrtf(ss), EPS);
    _Float16* dst = T + (size_t)n * DIM;
    for (int d = 0; d < DIM; d += 2) {
        union { _Float16 h[2]; unsigned u; } p;
        p.h[0] = (_Float16)(w[d * NCOL + n] * inv);
        p.h[1] = (_Float16)(w[(d + 1) * NCOL + n] * inv);
        *reinterpret_cast<unsigned*>(dst + d) = p.u;
    }
    rowS[n] = 0.0f;
}

// ---------------- Stage 2: symmetric Gram GEMM (f16 WMMA) + exp-sum ----------------
// Tiles jt >= it only; off-diagonal tiles also feed column sums (symmetry).
// B tiles stream global->LDS via async-to-LDS ops (ASYNCcnt), double-buffered one
// tile ahead per wave so a full tile of WMMA work hides the global latency.
// grid = (NTILE/2, JCHUNKS), block = 64 (2 waves). Wave = (iTile, chunk).
__global__ void __launch_bounds__(64)
gram_exp_kernel(const _Float16* __restrict__ T, float* __restrict__ rowS) {
    __shared__ __align__(16) _Float16 smem[2][2][16 * LROWH];  // [wave][buf][rows*paddedrow]

    const int wave = threadIdx.x >> 5;          // 0..1
    const int lane = threadIdx.x & 31;
    const int row  = lane & 15;                 // A-row / B-col owned by this lane
    const int hi   = lane >> 4;                 // half-wave select

    const int iTile = blockIdx.x * 2 + wave;    // 0..NTILE-1
    const int i0    = iTile * 16;
    const int chunk = blockIdx.y;               // 0..JCHUNKS-1

    // --- A fragments for the full K=256 (8 slices of K=32), resident in VGPRs. ---
    const _Float16* arow = T + (size_t)(i0 + row) * DIM;
    const int abase = hi * 16;
    half16 afrag[8];
#pragma unroll
    for (int t = 0; t < 8; ++t) {
        half8 lo = *(const half8*)(arow + t * 32 + abase);
        half8 hh = *(const half8*)(arow + t * 32 + 8 + abase);
        afrag[t] = __builtin_shufflevector(lo, hh, 0,1,2,3,4,5,6,7,8,9,10,11,12,13,14,15);
    }

    // Raw LDS byte offsets (addrspace(3)) for the async-copy destinations.
    typedef __attribute__((address_space(3))) _Float16 lds_half;
    const unsigned laneLds = (unsigned)(row * LROWB + hi * 256);
    unsigned ldsDst[2];
    ldsDst[0] = (unsigned)(size_t)(lds_half*)&smem[wave][0][0] + laneLds;
    ldsDst[1] = (unsigned)(size_t)(lds_half*)&smem[wave][1][0] + laneLds;

    // This lane's global byte offset within a tile (rows are contiguous 512B in T).
    const unsigned long long gLane =
        (unsigned long long)(size_t)T + (unsigned long long)(row * ROWB + hi * 256);

    // Issue 16 async b128 copies: one 8KB B tile (16 rows), this lane moves 256B.
    // INST_OFFSET is added to BOTH the LDS and global addresses (ISA 10.x async
    // addressing), so all 16 ops share one (lds, global) register pair and differ
    // only in the immediate offset -> no per-instruction address VALU.
    auto issue_copy = [&](int jt, int buf) {
        unsigned long long g = gLane + (unsigned long long)jt * (16u * ROWB);
        unsigned l = ldsDst[buf];
#define ACOPY(OFF) asm volatile("global_load_async_to_lds_b128 %0, %1, off offset:" OFF \
                                :: "v"(l), "v"(g) : "memory")
        ACOPY("0");   ACOPY("16");  ACOPY("32");  ACOPY("48");
        ACOPY("64");  ACOPY("80");  ACOPY("96");  ACOPY("112");
        ACOPY("128"); ACOPY("144"); ACOPY("160"); ACOPY("176");
        ACOPY("192"); ACOPY("208"); ACOPY("224"); ACOPY("240");
#undef ACOPY
    };

    float8 racc = {0.f,0.f,0.f,0.f,0.f,0.f,0.f,0.f};

    int jt = iTile + chunk;
    int buf = 0;
    if (jt < NTILE) issue_copy(jt, 0);          // prologue: first tile in flight

    for (; jt < NTILE; jt += JCHUNKS) {
        // Stage the NEXT tile (clamped dummy reload on the last iteration so the
        // outstanding count stays uniform).
        int pjt = jt + JCHUNKS; if (pjt >= NTILE) pjt = jt;
        issue_copy(pjt, buf ^ 1);

        // <=16 outstanding  =>  the current tile's 16 copies have completed.
        asm volatile("s_wait_asynccnt 0x10" ::: "memory");

        // B fragments from LDS: lanes0-15 K={k..k+15}, lanes16-31 K={k+16..k+31}.
        const _Float16* lrow = &smem[wave][buf][0] + row * LROWH + hi * 16;
        float8 c = {0.f,0.f,0.f,0.f,0.f,0.f,0.f,0.f};
#pragma unroll
        for (int t = 0; t < 8; ++t) {
            half16 bfrag = *(const half16*)(lrow + t * 32);   // 2x ds_load_b128
            c = __builtin_amdgcn_wmma_f32_16x16x32_f16(
                    false, afrag[t], false, bfrag, (short)0, c, false, false);
        }

        // exp(G_ij); row partials + per-lane column partial.
        float csum = 0.0f;
#pragma unroll
        for (int v = 0; v < 8; ++v) {
            float e = __builtin_amdgcn_exp2f(c[v] * LOG2E);
            racc[v] += e;
            csum += e;
        }

        // Symmetric contribution: tile column sums -> rows of block jt (skip diagonal).
        if (jt != iTile) {
            float cs = csum + __shfl_xor(csum, 16, 32);
            if (hi == 0)
                atomicAdd(&rowS[jt * 16 + row], cs);
        }
        buf ^= 1;
    }

    // Row sums: vgpr v, lanes0-15 -> row v, lanes16-31 -> row v+8; N = lane&15.
#pragma unroll
    for (int v = 0; v < 8; ++v) {
        float s = racc[v];
        s += __shfl_xor(s, 8, 16);
        s += __shfl_xor(s, 4, 16);
        s += __shfl_xor(s, 2, 16);
        s += __shfl_xor(s, 1, 16);
        if (row == 0)
            atomicAdd(&rowS[i0 + v + hi * 8], s);
    }
}

// ---------------- Stage 3: loss = lw * (sum_i log(S_i)/N - 1) ----------------
__global__ void finalize_kernel(const float* __restrict__ rowS,
                                const float* __restrict__ lw,
                                float* __restrict__ out) {
    __shared__ float red[256];
    float s = 0.0f;
    for (int n = threadIdx.x; n < NCOL; n += 256)
        s += __builtin_amdgcn_logf(rowS[n]);     // v_log_f32 is log2
    red[threadIdx.x] = s;
    __syncthreads();
    for (int k = 128; k > 0; k >>= 1) {
        if (threadIdx.x < (unsigned)k) red[threadIdx.x] += red[threadIdx.x + k];
        __syncthreads();
    }
    if (threadIdx.x == 0) {
        float sumlog = red[0] * LN2;             // log2 -> ln
        out[0] = lw[0] * (sumlog / (float)NCOL - 1.0f);
    }
}

extern "C" void kernel_launch(void* const* d_in, const int* in_sizes, int n_in,
                              void* d_out, int out_size, void* d_ws, size_t ws_size,
                              hipStream_t stream) {
    const float* w  = (const float*)d_in[0];     // [256, 16384] fp32
    const float* lw = (const float*)d_in[1];     // scalar fp32
    float* out = (float*)d_out;

    _Float16* T  = (_Float16*)d_ws;                                  // 8 MB packed f16
    float* rowS  = (float*)((char*)d_ws + (size_t)NCOL * DIM * 2);   // 64 KB row sums

    norm_pack_kernel<<<NCOL / 256, 256, 0, stream>>>(w, T, rowS);
    dim3 grid(NTILE / 2, JCHUNKS);               // 512 x 8 blocks, 2 waves each
    gram_exp_kernel<<<grid, 64, 0, stream>>>(T, rowS);
    finalize_kernel<<<1, 256, 0, stream>>>(rowS, lw, out);
}
